// TimeSeriesEncoder_75488345194628
// MI455X (gfx1250) — matrix operand order repaired and checked by reading, not compiled, not run on hardware
//
#include <hip/hip_runtime.h>
#include <math.h>

// ---------------- problem constants ----------------
#define BB     16
#define C_IN   64
#define LL     1024
#define HID    1024
#define OUTD   512
#define ROWS   (BB * LL)        // 16384
#define KD     3072             // 3 * HID (powers 1..3 concatenated)

// GEMM tiling
#define BMT    128              // block tile M
#define BNT    128              // block tile N
#define KSTEP  64               // K per stage (two wmma K-chunks)
#define ROWP   72               // LDS halves per row: 64 data + 8 pad (144B stride)
#define ATILE  (BMT * ROWP)     // 9216 halves
#define SSTR   (2 * ATILE)      // stage stride in halves (A tile + B tile)

typedef __attribute__((ext_vector_type(16))) __bf16        v16bf;
typedef __attribute__((ext_vector_type(8)))  float         v8f;
typedef __attribute__((ext_vector_type(4))) unsigned int   u32x4;
typedef __attribute__((ext_vector_type(4))) unsigned int   dx4u;
typedef __attribute__((ext_vector_type(8)))  int           dx8i;
typedef __attribute__((ext_vector_type(4)))  int           dx4i;

union frag_u { u32x4 u[2]; v16bf v; };

static __device__ __forceinline__ unsigned short f32_to_bf16(float f) {
    unsigned int u = __float_as_uint(f);
    u += 0x7FFFu + ((u >> 16) & 1u);        // round-to-nearest-even
    return (unsigned short)(u >> 16);
}

// Issue one TDM 2D tile load: tile (KSTEP x 128 rows) of bf16 from a row-major
// [nrows x kstride] tensor into LDS at lds_off, padding each 128B row to 144B.
// D# packing per CDNA5 ISA 8.3/8.4; groups 2-3 zero => 2D tensor.
static __device__ __forceinline__ void tdm_load_tile(const unsigned short* gsrc,
                                                     unsigned lds_off,
                                                     unsigned kstride,
                                                     unsigned nrows) {
    unsigned long long ga = (unsigned long long)(size_t)gsrc;
    dx4u g0 = {
        1u,                                               // count=1 (valid user D#)
        lds_off,                                          // lds_addr (bytes)
        (unsigned)(ga & 0xFFFFFFFFu),                     // global_addr[31:0]
        (unsigned)((ga >> 32) & 0x01FFFFFFu) | (2u << 30) // global_addr[56:32] | type=2
    };
    dx8i g1 = {
        (int)((1u << 16) | (1u << 20) | (4u << 22) | (3u << 25)), // 2B elems, pad_en, intv=32dw, amt=4dw
        (int)((kstride & 0xFFFFu) << 16),                         // tensor_dim0[15:0]
        (int)((kstride >> 16) | ((nrows & 0xFFFFu) << 16)),       // dim0[31:16] | dim1[15:0]
        (int)((nrows >> 16) | ((unsigned)KSTEP << 16)),           // dim1[31:16] | tile_dim0=64
        (int)BMT,                                                 // tile_dim1=128, tile_dim2=0
        (int)kstride,                                             // dim0_stride[31:0]
        0, 0
    };
    dx4i z4 = { 0, 0, 0, 0 };
    dx8i z8 = { 0, 0, 0, 0, 0, 0, 0, 0 };
    __builtin_amdgcn_tensor_load_to_lds(g0, g1, z4, z4, z8, 0);
}

// ---------------- K0: weight prep -------------------------------------------
__global__ void __launch_bounds__(256)
prep_w(const float* __restrict__ coeffs, const float* __restrict__ bias,
       unsigned short* __restrict__ WT, float* __restrict__ bp) {
    int o = blockIdx.x;
    int t = threadIdx.x;
    __shared__ float red[256];
    float s = 0.f;
    for (int c = t; c < HID; c += 256) {
        const float* cp = coeffs + ((size_t)o * HID + c) * 4;
        s += cp[0];
        unsigned short* w = WT + (size_t)o * KD + c;
        w[0]        = f32_to_bf16(cp[1]);
        w[HID]      = f32_to_bf16(cp[2]);
        w[2 * HID]  = f32_to_bf16(cp[3]);
    }
    red[t] = s; __syncthreads();
    for (int k = 128; k > 0; k >>= 1) { if (t < k) red[t] += red[t + k]; __syncthreads(); }
    if (t == 0) bp[o] = bias[o] + red[0];
}

// ---------------- K1: input projection + powers -----------------------------
__global__ void __launch_bounds__(256)
proj_pow(const float* __restrict__ x, const float* __restrict__ Win,
         const float* __restrict__ bin, unsigned short* __restrict__ PQ) {
    __shared__ float xs[64][65];
    __shared__ float ws[64][65];
    int t  = threadIdx.x;
    int r0 = blockIdx.y * 64;
    int j0 = blockIdx.x * 64;
    int b  = r0 >> 10;
    int l0 = r0 & 1023;
    for (int idx = t; idx < 64 * 64; idx += 256) {
        int c = idx >> 6, r = idx & 63;
        xs[c][r] = x[((size_t)b * C_IN + c) * LL + l0 + r];
        ws[c][r] = Win[(size_t)c * HID + j0 + r];
    }
    __syncthreads();
    int jj = t & 63;
    int rg = t >> 6;
    float bj = bin[j0 + jj];
    for (int rr = 0; rr < 16; ++rr) {
        int r = rg * 16 + rr;
        float acc = bj;
        #pragma unroll
        for (int c = 0; c < 64; ++c) acc += xs[c][r] * ws[c][jj];
        unsigned short* p = PQ + (size_t)(r0 + r) * KD + j0 + jj;
        float a2 = acc * acc;
        p[0]        = f32_to_bf16(acc);
        p[HID]      = f32_to_bf16(a2);
        p[2 * HID]  = f32_to_bf16(a2 * acc);
    }
}

// ---------------- K2: KAN GEMM, TDM triple-buffered LDS + WMMA --------------
// Y[row,n] = sum_k A[row,k]*BT[n,k] + bp[n].  A: ROWSxKD bf16, BT: NxKD bf16.
// Block: 8 waves, 128x128 tile; wave tile 32(M) x 64(N); 16 wmma per stage.
__global__ void __launch_bounds__(256)
kan_gemm(const unsigned short* __restrict__ A, const unsigned short* __restrict__ BT,
         const float* __restrict__ bp, float* __restrict__ Yo, int Nld) {
    __shared__ __align__(16) unsigned short smem[3][SSTR];  // 3 x (A tile + B tile)

    int lane = threadIdx.x & 31;
    int w    = threadIdx.x >> 5;
    int mw   = w & 3;                       // 0..3 : M sub-tile
    int nw   = w >> 2;                      // 0..1 : N sub-tile
    int lr   = lane & 15;
    int lg   = lane >> 4;

    const unsigned short* Ab = A  + (size_t)(blockIdx.y * BMT) * KD;
    const unsigned short* Bb = BT + (size_t)(blockIdx.x * BNT) * KD;
    unsigned short* sf = &smem[0][0];
    unsigned lds0 = (unsigned)(size_t)sf;   // LDS byte base of buffers

    v8f acc[2][4];
    v8f zz = {};
    #pragma unroll
    for (int i = 0; i < 2; ++i)
        #pragma unroll
        for (int j = 0; j < 4; ++j) acc[i][j] = zz;

    const int steps = KD / KSTEP;           // 48
    if (w == 0) {                           // prime stages 0 and 1
        tdm_load_tile(Ab,                 lds0,                       KD, ROWS);
        tdm_load_tile(Bb,                 lds0 + 2u * ATILE,          KD, (unsigned)Nld);
        tdm_load_tile(Ab + KSTEP,         lds0 + 2u * SSTR,           KD, ROWS);
        tdm_load_tile(Bb + KSTEP,         lds0 + 2u * (SSTR + ATILE), KD, (unsigned)Nld);
        __builtin_amdgcn_s_wait_tensorcnt(2);   // stage 0 landed (in-order)
    }
    __syncthreads();

    unsigned bc = 0, b1 = SSTR, b2 = 2 * SSTR;  // element offsets: cur / next / filling
    #pragma unroll 1
    for (int s = 0; s < steps; ++s) {
        if (w == 0 && s + 2 < steps) {      // DMA stage s+2 while computing s
            tdm_load_tile(Ab + (size_t)(s + 2) * KSTEP, lds0 + 2u * b2,            KD, ROWS);
            tdm_load_tile(Bb + (size_t)(s + 2) * KSTEP, lds0 + 2u * (b2 + ATILE),  KD, (unsigned)Nld);
        }

        #pragma unroll
        for (int c = 0; c < 2; ++c) {       // two 32-wide K chunks per stage
            frag_u fa[2];
            #pragma unroll
            for (int i = 0; i < 2; ++i) {   // A: lane-group = low/high 8 of each 16-K chunk
                unsigned ar = bc + (mw * 32 + i * 16 + lr) * ROWP + c * 32 + lg * 8;
                fa[i].u[0] = *(const u32x4*)&sf[ar];
                fa[i].u[1] = *(const u32x4*)&sf[ar + 16];
            }
            #pragma unroll
            for (int j = 0; j < 4; ++j) {   // B: lane-group = low/high 8 of each 16-K chunk
                unsigned br = bc + ATILE + (nw * 64 + j * 16 + lr) * ROWP + c * 32 + lg * 8;
                frag_u fb;
                fb.u[0] = *(const u32x4*)&sf[br];
                fb.u[1] = *(const u32x4*)&sf[br + 16];
                acc[0][j] = __builtin_amdgcn_wmma_f32_16x16x32_bf16(
                    false, fa[0].v, false, fb.v, (short)0, acc[0][j], false, false);
                acc[1][j] = __builtin_amdgcn_wmma_f32_16x16x32_bf16(
                    false, fa[1].v, false, fb.v, (short)0, acc[1][j], false, false);
            }
        }

        if (w == 0) {                       // ensure stage s+1 landed; s+2 stays in flight
            if (s + 2 < steps) __builtin_amdgcn_s_wait_tensorcnt(2);
            else               __builtin_amdgcn_s_wait_tensorcnt(0);
        }
        __syncthreads();
        unsigned t0 = bc; bc = b1; b1 = b2; b2 = t0;   // rotate ring
    }

    // C/D layout: VGPR r -> M = r + 8*laneGroup; N = lane%16
    #pragma unroll
    for (int i = 0; i < 2; ++i) {
        #pragma unroll
        for (int j = 0; j < 4; ++j) {
            int n = blockIdx.x * BNT + nw * 64 + j * 16 + lr;
            float bb = bp[n];
            #pragma unroll
            for (int r = 0; r < 8; ++r) {
                int m = blockIdx.y * BMT + mw * 32 + i * 16 + lg * 8 + r;
                Yo[(size_t)m * Nld + n] = acc[i][j][r] + bb;
            }
        }
    }
}

// ---------------- K3: LayerNorm + exact GELU + powers -----------------------
__global__ void __launch_bounds__(256)
ln_gelu_pow(const float* __restrict__ Y, const float* __restrict__ g,
            const float* __restrict__ be, unsigned short* __restrict__ PQ) {
    int row = blockIdx.x, t = threadIdx.x;
    __shared__ float rs[256], rs2[256];
    const float* y = Y + (size_t)row * HID;
    float v[4], s = 0.f, s2 = 0.f;
    #pragma unroll
    for (int i = 0; i < 4; ++i) { v[i] = y[t + i * 256]; s += v[i]; s2 += v[i] * v[i]; }
    rs[t] = s; rs2[t] = s2; __syncthreads();
    for (int k = 128; k > 0; k >>= 1) {
        if (t < k) { rs[t] += rs[t + k]; rs2[t] += rs2[t + k]; }
        __syncthreads();
    }
    float mu   = rs[0] * (1.f / HID);
    float var  = rs2[0] * (1.f / HID) - mu * mu;
    float rstd = rsqrtf(var + 1e-5f);
    unsigned short* p = PQ + (size_t)row * KD;
    #pragma unroll
    for (int i = 0; i < 4; ++i) {
        int j = t + i * 256;
        float u  = (v[i] - mu) * rstd * g[j] + be[j];
        float ge = 0.5f * u * (1.f + erff(u * 0.70710678118654752f));
        float g2 = ge * ge;
        p[j]           = f32_to_bf16(ge);
        p[j + HID]     = f32_to_bf16(g2);
        p[j + 2 * HID] = f32_to_bf16(g2 * ge);
    }
}

// ---------------- K4: final LayerNorm (in place, width 512) -----------------
__global__ void __launch_bounds__(256)
ln2_inplace(float* __restrict__ Y, const float* __restrict__ g, const float* __restrict__ be) {
    int row = blockIdx.x, t = threadIdx.x;
    __shared__ float rs[256], rs2[256];
    float* y = Y + (size_t)row * OUTD;
    float v0 = y[t], v1 = y[t + 256];
    rs[t] = v0 + v1; rs2[t] = v0 * v0 + v1 * v1; __syncthreads();
    for (int k = 128; k > 0; k >>= 1) {
        if (t < k) { rs[t] += rs[t + k]; rs2[t] += rs2[t + k]; }
        __syncthreads();
    }
    float mu   = rs[0] * (1.f / OUTD);
    float var  = rs2[0] * (1.f / OUTD) - mu * mu;
    float rstd = rsqrtf(var + 1e-5f);
    y[t]       = (v0 - mu) * rstd * g[t] + be[t];
    y[t + 256] = (v1 - mu) * rstd * g[t + 256] + be[t + 256];
}

// ---------------- K5: mean pool over L (deterministic fixed-order sum) ------
__global__ void __launch_bounds__(256)
pool_mean(const float* __restrict__ Y, float* __restrict__ out) {
    int idx = blockIdx.x * 256 + threadIdx.x;
    int b = idx >> 9, o = idx & 511;
    const float* y = Y + ((size_t)b * LL) * OUTD + o;
    float s = 0.f;
    for (int l = 0; l < LL; ++l) s += y[(size_t)l * OUTD];
    out[idx] = s * (1.f / LL);
}

// ---------------- launcher --------------------------------------------------
extern "C" void kernel_launch(void* const* d_in, const int* in_sizes, int n_in,
                              void* d_out, int out_size, void* d_ws, size_t ws_size,
                              hipStream_t stream) {
    const float* x       = (const float*)d_in[0];
    const float* W_in    = (const float*)d_in[1];
    const float* b_in    = (const float*)d_in[2];
    const float* coeffs1 = (const float*)d_in[3];
    const float* bias1   = (const float*)d_in[4];
    const float* g1      = (const float*)d_in[5];
    const float* beta1   = (const float*)d_in[6];
    const float* coeffs2 = (const float*)d_in[7];
    const float* bias2   = (const float*)d_in[8];
    const float* g2      = (const float*)d_in[9];
    const float* beta2   = (const float*)d_in[10];
    float* out = (float*)d_out;

    char* ws = (char*)d_ws;
    size_t offPQ = 0;                                         // bf16 16384x3072
    size_t offY  = offPQ + (size_t)ROWS * KD * 2;             // f32  16384x1024 (reused 16384x512)
    size_t offW1 = offY  + (size_t)ROWS * HID * 4;            // bf16 1024x3072
    size_t offW2 = offW1 + (size_t)HID * KD * 2;              // bf16 512x3072
    size_t offB1 = offW2 + (size_t)OUTD * KD * 2;             // f32  1024
    size_t offB2 = offB1 + (size_t)HID * 4;                   // f32  512
    unsigned short* PQ  = (unsigned short*)(ws + offPQ);
    float*          Y   = (float*)(ws + offY);
    unsigned short* W1T = (unsigned short*)(ws + offW1);
    unsigned short* W2T = (unsigned short*)(ws + offW2);
    float*          b1p = (float*)(ws + offB1);
    float*          b2p = (float*)(ws + offB2);

    prep_w<<<HID,  256, 0, stream>>>(coeffs1, bias1, W1T, b1p);
    prep_w<<<OUTD, 256, 0, stream>>>(coeffs2, bias2, W2T, b2p);

    proj_pow<<<dim3(HID / 64, ROWS / 64), 256, 0, stream>>>(x, W_in, b_in, PQ);

    kan_gemm<<<dim3(HID / BNT, ROWS / BMT), 256, 0, stream>>>(PQ, W1T, b1p, Y, HID);

    ln_gelu_pow<<<ROWS, 256, 0, stream>>>(Y, g1, beta1, PQ);

    kan_gemm<<<dim3(OUTD / BNT, ROWS / BMT), 256, 0, stream>>>(PQ, W2T, b2p, Y, OUTD);

    ln2_inplace<<<ROWS, 256, 0, stream>>>(Y, g2, beta2);

    pool_mean<<<(BB * OUTD) / 256, 256, 0, stream>>>(Y, out);
}
